// MSGCN_37340445671874
// MI455X (gfx1250) — matrix operand order repaired and checked by reading, not compiled, not run on hardware
//
#include <hip/hip_runtime.h>
#include <hip/hip_bf16.h>

typedef __attribute__((ext_vector_type(16))) __bf16 v16bf;
typedef __attribute__((ext_vector_type(8)))  float  v8f;
typedef __attribute__((ext_vector_type(4)))  unsigned int v4u;
typedef __attribute__((ext_vector_type(8)))  int v8i;
typedef __attribute__((ext_vector_type(4)))  int v4i;

#define N_TOTAL 65536       // B * N_PER
#define E_TOTAL 1048576     // B * E_PER
#define E_MASK  2047        // E_PER - 1 (w = tile(edge_weight, B))
#define WAVES   8

#if defined(__has_builtin)
#if __has_builtin(__builtin_amdgcn_tensor_load_to_lds)
#define HAVE_TDM 1
#endif
#endif
#ifndef HAVE_TDM
#define HAVE_TDM 0
#endif

// amdgpu-toolchain (therock headers) => 6-arg builtin; ROCm 7.2 => 5-arg.
#if __has_include(<hip/amd_detail/amd_gfx1250_TDM.h>)
#define TDM_6ARG 1
#else
#define TDM_6ARG 0
#endif

__device__ __forceinline__ float celu1(float v) {
    return v > 0.0f ? v : (__expf(v) - 1.0f);
}

#if HAVE_TDM
// Issue a TDM load of a 32(K-rows) x 16(N-cols) fp32 tile into LDS.
// D# group0: count=1 | lds_addr | global tile addr | type=2.
// D# group1: data_size=4B, tensor_dim0=ldb, tensor_dim1=K, tile 16x32, stride0=ldb.
__device__ __forceinline__ void tdm_load_tile(const float* gptr, unsigned ldsAddr,
                                              int ldb, int Ktot) {
    unsigned long long ga = (unsigned long long)(uintptr_t)gptr;
    v4u g0;
    g0[0] = 1u;                                            // count=1, user mode
    g0[1] = ldsAddr;                                       // lds_addr (bytes)
    g0[2] = (unsigned)ga;                                  // global_addr[31:0]
    g0[3] = (unsigned)((ga >> 32) & 0x01FFFFFFu) | 0x80000000u; // addr[56:32] | type=2
    v8i g1;
    g1[0] = 0x20000;                                       // data_size=2 (4 bytes)
    g1[1] = (int)((unsigned)(ldb & 0xFFFF) << 16);         // tensor_dim0[15:0]
    g1[2] = (int)(((unsigned)ldb >> 16) & 0xFFFFu);        // tensor_dim0[31:16], tdim1 lo=K
    g1[2] |= (int)((unsigned)(Ktot & 0xFFFF) << 16);
    g1[3] = (int)(((unsigned)Ktot >> 16) & 0xFFFFu) | (16 << 16); // tile_dim0=16
    g1[4] = 32;                                            // tile_dim1=32, tile_dim2=0
    g1[5] = ldb;                                           // tensor_dim0_stride[31:0]
    g1[6] = 0;
    g1[7] = 0;
    v4i z4 = {0, 0, 0, 0};
#if TDM_6ARG
    v8i z8 = {0, 0, 0, 0, 0, 0, 0, 0};
    __builtin_amdgcn_tensor_load_to_lds(g0, g1, z4, z4, z8, 0);
#else
    __builtin_amdgcn_tensor_load_to_lds(g0, g1, z4, z4, 0);
#endif
}
#endif

// ---------------------------------------------------------------------------
// BatchNorm statistics: two-pass deterministic reduction.
// ---------------------------------------------------------------------------
__global__ void bn_partial_kernel(const float* __restrict__ x,
                                  float* __restrict__ psum, float* __restrict__ psqs,
                                  int rowsPerBlk) {
    int f = threadIdx.x;                  // 128 features
    int r0 = blockIdx.x * rowsPerBlk;
    float s = 0.0f, q = 0.0f;
    for (int r = 0; r < rowsPerBlk; ++r) {
        float v = x[(size_t)(r0 + r) * 128 + f];
        s += v; q += v * v;
    }
    psum[blockIdx.x * 128 + f] = s;
    psqs[blockIdx.x * 128 + f] = q;
}

__global__ void bn_final_kernel(const float* __restrict__ psum, const float* __restrict__ psqs,
                                int nblk, int n,
                                float* __restrict__ mean, float* __restrict__ rstd) {
    int f = threadIdx.x;
    float s = 0.0f, q = 0.0f;
    for (int b = 0; b < nblk; ++b) { s += psum[b * 128 + f]; q += psqs[b * 128 + f]; }
    float m = s / (float)n;
    float var = q / (float)n - m * m;     // biased variance, matches reference
    mean[f] = m;
    rstd[f] = rsqrtf(var + 1e-5f);
}

// ---------------------------------------------------------------------------
// Degree / dinv
// ---------------------------------------------------------------------------
__global__ void deg_init_kernel(float* __restrict__ deg, int n) {
    int i = blockIdx.x * blockDim.x + threadIdx.x;
    if (i < n) deg[i] = 1.0f;             // +1 self-loop weight
}

__global__ void deg_scatter_kernel(const int* __restrict__ col, const float* __restrict__ ew,
                                   float* __restrict__ deg, int nEdges) {
    int e = blockIdx.x * blockDim.x + threadIdx.x;
    if (e < nEdges)
        __hip_atomic_fetch_add(&deg[col[e]], ew[e & E_MASK],
                               __ATOMIC_RELAXED, __HIP_MEMORY_SCOPE_AGENT);
}

__global__ void dinv_kernel(const float* __restrict__ deg, float* __restrict__ dinv, int n) {
    int i = blockIdx.x * blockDim.x + threadIdx.x;
    if (i < n) dinv[i] = rsqrtf(deg[i]);
}

// ---------------------------------------------------------------------------
// WMMA GEMM: D[M,N] = preop(A)[M,K] @ B[K,N]  (bf16 in, f32 accumulate)
// PRE: 0=none, 1=batchnorm(per-K), 2=celu.  POST: 0=raw store, 1=celu(v+bias)
// Block = 8 waves; wave w owns rows [blockIdx.y*128 + 16w, +16), cols [16*blockIdx.x, +16).
// B tiles staged to LDS by the Tensor Data Mover (double-buffered, wave 0 issues,
// s_wait_tensorcnt fences); A fragments loaded as float2 with fused pre-op.
// ---------------------------------------------------------------------------
template <int PRE, int POST>
__global__ __launch_bounds__(256)
void wmma_gemm_kernel(const float* __restrict__ A, const float* __restrict__ B,
                      float* __restrict__ D,
                      int M, int N, int K, int lda, int ldb, int ldd,
                      const float* __restrict__ mean, const float* __restrict__ rstd,
                      const float* __restrict__ gamma, const float* __restrict__ beta,
                      const float* __restrict__ bias) {
    __shared__ float sB[2][32 * 16];      // double-buffered 32(K) x 16(N) fp32 tile, 4 KB

    const int lane = threadIdx.x & 31;
    const int wave = threadIdx.x >> 5;
    const int half = lane >> 4;           // lane 16-31 -> 1
    const int l16  = lane & 15;
    const int n0   = blockIdx.x * 16;
    const int m0   = blockIdx.y * (16 * WAVES) + wave * 16;
    const int mrow = m0 + l16;            // A-fragment row for this lane
    const int ncol = n0 + l16;            // D column for this lane

#if HAVE_TDM
    // Prologue: DMA tile k0=0 into buffer 0.
    if (wave == 0) {
        tdm_load_tile(B + (size_t)0 * ldb + n0,
                      (unsigned)(uintptr_t)&sB[0][0], ldb, K);
        __builtin_amdgcn_s_wait_tensorcnt(0);
    }
    __syncthreads();
#endif

    v8f acc = {};

    for (int k0 = 0; k0 < K; k0 += 32) {
        const int cur = (k0 >> 5) & 1;
#if HAVE_TDM
        // Kick off DMA of the next tile while we compute on this one.
        if (wave == 0 && (k0 + 32) < K)
            tdm_load_tile(B + (size_t)(k0 + 32) * ldb + n0,
                          (unsigned)(uintptr_t)&sB[cur ^ 1][0], ldb, K);
#else
        for (int t = threadIdx.x; t < 32 * 16; t += 256) {
            int kk = t >> 4, nn = t & 15;
            sB[cur][t] = B[(size_t)(k0 + kk) * ldb + (n0 + nn)];
        }
        __syncthreads();
#endif

        // B fragment: element e -> K = 8*half + (e&7) + 16*(e>>3), N = l16.
        v16bf bfrag;
#pragma unroll
        for (int e = 0; e < 16; ++e)
            bfrag[e] = (__bf16)sB[cur][((half << 3) + (e & 7) + ((e >> 3) << 4)) * 16 + l16];

        // A fragment with fused pre-op; CDNA5 16-bit A layout, float2 loads.
        v16bf afrag;
        const float* Arow = A + (size_t)mrow * lda + k0 + (half ? 8 : 0);
        if ((k0 + 32) < K) __builtin_prefetch(Arow + 32, 0, 0);
#pragma unroll
        for (int v = 0; v < 8; ++v) {
            const int kk = ((v >= 4) ? 16 : 0) + ((v & 3) << 1);   // K pair base
            float2 pr = *(const float2*)(Arow + kk);
            float px = pr.x, py = pr.y;
            if (PRE == 1) {
                int kg = k0 + (half ? 8 : 0) + kk;
                px = (px - mean[kg])     * rstd[kg]     * gamma[kg]     + beta[kg];
                py = (py - mean[kg + 1]) * rstd[kg + 1] * gamma[kg + 1] + beta[kg + 1];
            } else if (PRE == 2) {
                px = celu1(px); py = celu1(py);
            }
            afrag[2 * v]     = (__bf16)px;
            afrag[2 * v + 1] = (__bf16)py;
        }

        acc = __builtin_amdgcn_wmma_f32_16x16x32_bf16(
            /*neg_a=*/false, afrag, /*neg_b=*/false, bfrag,
            /*c_mod=*/(short)0, acc, /*reuse_a=*/false, /*reuse_b=*/false);

#if HAVE_TDM
        if (wave == 0) __builtin_amdgcn_s_wait_tensorcnt(0);
#endif
        __syncthreads();
    }

    // D store: VGPR r -> row m0 + r + 8*half, col ncol.
#pragma unroll
    for (int r = 0; r < 8; ++r) {
        int mr = m0 + r + half * 8;
        float v = acc[r];
        if (POST == 1) v = celu1(v + bias[ncol]);
        D[(size_t)mr * ldd + ncol] = v;
    }
}

// ---------------------------------------------------------------------------
// Self-loop init: h = xw * dinv^2 + b   (the d^-1/2 * 1 * d^-1/2 term + bias)
// ---------------------------------------------------------------------------
__global__ void init_h_kernel(const float* __restrict__ xw, const float* __restrict__ dinv,
                              const float* __restrict__ b, float* __restrict__ h,
                              int F, int n) {
    size_t idx = (size_t)blockIdx.x * blockDim.x + threadIdx.x;
    if (idx >= (size_t)n * F) return;
    int i = (int)(idx / F), f = (int)(idx % F);
    float di = dinv[i];
    h[idx] = xw[idx] * di * di + b[f];
}

// ---------------------------------------------------------------------------
// Weighted scatter-add: h[col] += dinv[row]*w*dinv[col] * xw[row]
// One wave per edge (wave32), lanes stride the feature dim; native f32 atomics.
// ---------------------------------------------------------------------------
__global__ void edge_scatter_kernel(const float* __restrict__ xw, const int* __restrict__ row,
                                    const int* __restrict__ col, const float* __restrict__ ew,
                                    const float* __restrict__ dinv, float* __restrict__ out,
                                    int F, int nEdges) {
    int e = blockIdx.x * (blockDim.x >> 5) + (threadIdx.x >> 5);
    if (e >= nEdges) return;
    int lane = threadIdx.x & 31;
    int r = row[e], c = col[e];
    float norm = dinv[r] * ew[e & E_MASK] * dinv[c];
    const float* src = xw + (size_t)r * F;
    float* dst = out + (size_t)c * F;
    for (int f = lane; f < F; f += 32)
        __hip_atomic_fetch_add(&dst[f], norm * src[f],
                               __ATOMIC_RELAXED, __HIP_MEMORY_SCOPE_AGENT);
}

// ---------------------------------------------------------------------------
// FC2: out[512,8] = H[512,1024] @ W[1024,8] + b   (tiny, plain FMA loop)
// ---------------------------------------------------------------------------
__global__ void fc2_kernel(const float* __restrict__ H, const float* __restrict__ W,
                           const float* __restrict__ b, float* __restrict__ out) {
    int idx = blockIdx.x * blockDim.x + threadIdx.x;   // 4096 outputs
    if (idx >= 512 * 8) return;
    int bi = idx >> 3, o = idx & 7;
    const float* h = H + (size_t)bi * 1024;
    float s = b[o];
    for (int k = 0; k < 1024; ++k) s = fmaf(h[k], W[k * 8 + o], s);
    out[idx] = s;
}

// ---------------------------------------------------------------------------
extern "C" void kernel_launch(void* const* d_in, const int* in_sizes, int n_in,
                              void* d_out, int out_size, void* d_ws, size_t ws_size,
                              hipStream_t stream) {
    const float* x    = (const float*)d_in[0];
    const int*   eidx = (const int*)  d_in[1];
    const float* ew   = (const float*)d_in[2];
    const float* gam  = (const float*)d_in[3];
    const float* bet  = (const float*)d_in[4];
    const float* W1   = (const float*)d_in[5];
    const float* b1   = (const float*)d_in[6];
    const float* W2   = (const float*)d_in[7];
    const float* b2   = (const float*)d_in[8];
    const float* fcW1 = (const float*)d_in[9];
    const float* fcb1 = (const float*)d_in[10];
    const float* fcW2 = (const float*)d_in[11];
    const float* fcb2 = (const float*)d_in[12];
    float* out = (float*)d_out;

    const int* row = eidx;            // edge_index[0]
    const int* col = eidx + E_TOTAL;  // edge_index[1]

    // Bump allocator over d_ws (~99 MB total, fits L2).
    char* p = (char*)d_ws;
    auto alloc = [&](size_t bytes) -> float* {
        float* r = (float*)p;
        p += (bytes + 255) & ~(size_t)255;
        return r;
    };
    float* mean = alloc(128 * 4);
    float* rstd = alloc(128 * 4);
    float* psum = alloc(256 * 128 * 4);
    float* psqs = alloc(256 * 128 * 4);
    float* deg  = alloc((size_t)N_TOTAL * 4);
    float* dinv = alloc((size_t)N_TOTAL * 4);
    float* xw1  = alloc((size_t)N_TOTAL * 128 * 4);
    float* h1   = alloc((size_t)N_TOTAL * 128 * 4);
    float* xw2  = alloc((size_t)N_TOTAL * 64 * 4);
    float* h2   = alloc((size_t)N_TOTAL * 64 * 4);
    float* h3   = alloc((size_t)512 * 1024 * 4);

    // BatchNorm statistics (deterministic two-pass).
    bn_partial_kernel<<<256, 128, 0, stream>>>(x, psum, psqs, 256);
    bn_final_kernel<<<1, 128, 0, stream>>>(psum, psqs, 256, N_TOTAL, mean, rstd);

    // Degrees and dinv.
    deg_init_kernel<<<N_TOTAL / 256, 256, 0, stream>>>(deg, N_TOTAL);
    deg_scatter_kernel<<<E_TOTAL / 256, 256, 0, stream>>>(col, ew, deg, E_TOTAL);
    dinv_kernel<<<N_TOTAL / 256, 256, 0, stream>>>(deg, dinv, N_TOTAL);

    // Layer 1: xw1 = BN(x) @ W1 ; h1 = xw1*dinv^2 + b1 ; scatter.
    wmma_gemm_kernel<1, 0><<<dim3(128 / 16, N_TOTAL / 128), 256, 0, stream>>>(
        x, W1, xw1, N_TOTAL, 128, 128, 128, 128, 128, mean, rstd, gam, bet, nullptr);
    init_h_kernel<<<(N_TOTAL * 128) / 256, 256, 0, stream>>>(xw1, dinv, b1, h1, 128, N_TOTAL);
    edge_scatter_kernel<<<E_TOTAL / 8, 256, 0, stream>>>(xw1, row, col, ew, dinv, h1, 128, E_TOTAL);

    // Layer 2: xw2 = celu(h1) @ W2 ; h2 = xw2*dinv^2 + b2 ; scatter.
    wmma_gemm_kernel<2, 0><<<dim3(64 / 16, N_TOTAL / 128), 256, 0, stream>>>(
        h1, W2, xw2, N_TOTAL, 64, 128, 128, 64, 64, nullptr, nullptr, nullptr, nullptr, nullptr);
    init_h_kernel<<<(N_TOTAL * 64) / 256, 256, 0, stream>>>(xw2, dinv, b2, h2, 64, N_TOTAL);
    edge_scatter_kernel<<<E_TOTAL / 8, 256, 0, stream>>>(xw2, row, col, ew, dinv, h2, 64, E_TOTAL);

    // FC1: h3 = celu( celu(h2).reshape(512,8192) @ fcW1 + fcb1 )  — reshape is a no-op view.
    wmma_gemm_kernel<2, 1><<<dim3(1024 / 16, 512 / 128), 256, 0, stream>>>(
        (const float*)h2, fcW1, h3, 512, 1024, 8192, 8192, 1024, 1024,
        nullptr, nullptr, nullptr, nullptr, fcb1);

    // FC2 logits.
    fc2_kernel<<<16, 256, 0, stream>>>(h3, fcW2, fcb2, out);
}
